// InformerCore_53901839565143
// MI455X (gfx1250) — matrix operand order repaired and checked by reading, not compiled, not run on hardware
//
#include <hip/hip_runtime.h>
#include <cmath>
#include <cstdint>
#include <cstddef>

// ---------------------------------------------------------------------------
// Types for CDNA5 WMMA (wave32)
// ---------------------------------------------------------------------------
typedef _Float16 h16;
typedef __attribute__((ext_vector_type(16))) _Float16 v16h;
typedef __attribute__((ext_vector_type(8)))  _Float16 v8h;
typedef __attribute__((ext_vector_type(8)))  float    v8f;
typedef __attribute__((ext_vector_type(4)))  float    v4f;

union FragH { v16h v; v8h h8[2]; h16 e[16]; };

static __device__ inline v8f wmma_f16(v16h a, v16h b, v8f c) {
  // v_wmma_f32_16x16x32_f16  (8-arg form, probe-confirmed)
  return __builtin_amdgcn_wmma_f32_16x16x32_f16(
      /*neg_a=*/false, a, /*neg_b=*/false, b,
      /*c_mod=*/(short)0, c, /*reuse_a=*/false, /*reuse_b=*/false);
}

static __device__ inline float gelu_exact(float x) {
  return 0.5f * x * (1.0f + erff(x * 0.70710678118654752f));
}

// ---------------------------------------------------------------------------
// FAST WMMA GEMM (no bounds checks): requires M%64==0, N%128==0, K%32==0.
// C[M,N] = A[M,K](f16) * Wt[N,K](f16)^T + bias [+ residual] [GELU]
// block = 128 threads (4 wave32s, 2x2 wave grid), block tile 64(M) x 128(N),
// wave tile 32x64 = 2x4 fragments = 8 v_wmma per K-step.
// A frag (ISA 7.12.2): lane<16 -> row=lane, K chunks [k0,k0+8)+[k0+16,k0+24);
//                      lane>=16 -> same row set, chunks shifted +8.
// B frag: col=lane%16, same K chunks as A frag (lane>=16 shifted +8).
// Bw is weights [N,K] row-major => B column n == contiguous W row n.
// ---------------------------------------------------------------------------
__global__ void __launch_bounds__(128)
k_gemm_wmma_fast(const h16* __restrict__ A, const h16* __restrict__ Bw,
                 const float* __restrict__ bias, const float* __restrict__ residual,
                 float* __restrict__ C, int M, int N, int K, int act) {
  int wave = threadIdx.x >> 5;
  int lane = threadIdx.x & 31;
  int m_base = blockIdx.y * 64  + (wave >> 1) * 32;
  int n_base = blockIdx.x * 128 + (wave & 1) * 64;

  // per-lane base pointers (constant across K loop)
  const h16* pa0 = A  + (size_t)(m_base + (lane & 15)) * K + ((lane >> 4) << 3);
  const h16* pa1 = pa0 + (size_t)16 * K;
  const h16* pb  = Bw + (size_t)(n_base + (lane & 15)) * K + ((lane >> 4) << 3);
  const size_t colstep = (size_t)16 * K;

  v8f acc[2][4];
#pragma unroll
  for (int i = 0; i < 2; ++i)
#pragma unroll
    for (int j = 0; j < 4; ++j) acc[i][j] = (v8f)0.0f;

  for (int k0 = 0; k0 < K; k0 += 32) {
    // speculative prefetch of next K-tile (CDNA5 global_prefetch_b8)
    __builtin_prefetch(pa0 + k0 + 32, 0, 1);
    __builtin_prefetch(pa1 + k0 + 32, 0, 1);
    __builtin_prefetch(pb  + k0 + 32, 0, 1);

    FragH a0, a1;
    a0.h8[0] = *(const v8h*)(pa0 + k0);
    a0.h8[1] = *(const v8h*)(pa0 + k0 + 16);
    a1.h8[0] = *(const v8h*)(pa1 + k0);
    a1.h8[1] = *(const v8h*)(pa1 + k0 + 16);

    FragH b[4];
#pragma unroll
    for (int ni = 0; ni < 4; ++ni) {
      const h16* p = pb + (size_t)ni * colstep + k0;
      b[ni].h8[0] = *(const v8h*)p;
      b[ni].h8[1] = *(const v8h*)(p + 16);
    }

#pragma unroll
    for (int ni = 0; ni < 4; ++ni) {
      acc[0][ni] = wmma_f16(a0.v, b[ni].v, acc[0][ni]);
      acc[1][ni] = wmma_f16(a1.v, b[ni].v, acc[1][ni]);
    }
  }

  // C/D layout: VGPR v -> M = v + 8*(lane>=16); N = lane % 16.
  int mofs = (lane >> 4) << 3;
  int ncol = lane & 15;
#pragma unroll
  for (int mi = 0; mi < 2; ++mi) {
#pragma unroll
    for (int ni = 0; ni < 4; ++ni) {
      int n = n_base + ni * 16 + ncol;
#pragma unroll
      for (int v = 0; v < 8; ++v) {
        int m = m_base + mi * 16 + v + mofs;
        float val = acc[mi][ni][v];
        if (bias) val += bias[n];
        if (residual) val += residual[(size_t)m * N + n];
        if (act == 1) val = gelu_exact(val);
        C[(size_t)m * N + n] = val;
      }
    }
  }
}

// ---------------------------------------------------------------------------
// Generic (edge-case) WMMA GEMM with bounds checks: block tile 64x64.
// Used only for ragged shapes (K=63 embeddings, N=21 projection).
// ---------------------------------------------------------------------------
static __device__ inline v16h load_frag_a(const h16* __restrict__ A, int M, int K,
                                          int row0, int k0, int lane) {
  FragH f;
  int r  = row0 + (lane & 15);
  int kb = k0 + ((lane >> 4) << 3);
  bool fast = (r < M) && ((K & 7) == 0) && (kb + 24 <= K);
  if (fast) {
    const h16* p = A + (size_t)r * K + kb;
    f.h8[0] = *(const v8h*)p;
    f.h8[1] = *(const v8h*)(p + 16);
  } else {
#pragma unroll
    for (int i = 0; i < 8; ++i) {
      int k1 = kb + i, k2 = kb + 16 + i;
      f.e[i]     = (r < M && k1 < K) ? A[(size_t)r * K + k1] : (h16)0;
      f.e[8 + i] = (r < M && k2 < K) ? A[(size_t)r * K + k2] : (h16)0;
    }
  }
  return f.v;
}

static __device__ inline v16h load_frag_b(const h16* __restrict__ Bw, int N, int K,
                                          int col0, int k0, int lane) {
  FragH f;
  int c  = col0 + (lane & 15);
  int kb = k0 + ((lane >> 4) << 3);
  bool fast = (c < N) && ((K & 7) == 0) && (kb + 24 <= K);
  if (fast) {
    const h16* p = Bw + (size_t)c * K + kb;
    f.h8[0] = *(const v8h*)p;
    f.h8[1] = *(const v8h*)(p + 16);
  } else {
#pragma unroll
    for (int i = 0; i < 16; ++i) {
      int kk = kb + (i < 8 ? i : i + 8);
      f.e[i] = (c < N && kk < K) ? Bw[(size_t)c * K + kk] : (h16)0;
    }
  }
  return f.v;
}

__global__ void __launch_bounds__(128)
k_gemm_wmma_gen(const h16* __restrict__ A, const h16* __restrict__ Bw,
                const float* __restrict__ bias, const float* __restrict__ residual,
                float* __restrict__ C, int M, int N, int K, int act) {
  int wave = threadIdx.x >> 5;
  int lane = threadIdx.x & 31;
  int m_base = blockIdx.y * 64 + (wave >> 1) * 32;
  int n_base = blockIdx.x * 64 + (wave & 1) * 32;

  v8f acc[2][2];
#pragma unroll
  for (int i = 0; i < 2; ++i)
#pragma unroll
    for (int j = 0; j < 2; ++j) acc[i][j] = (v8f)0.0f;

  for (int k0 = 0; k0 < K; k0 += 32) {
    v16h a0 = load_frag_a(A, M, K, m_base,      k0, lane);
    v16h a1 = load_frag_a(A, M, K, m_base + 16, k0, lane);
    v16h b0 = load_frag_b(Bw, N, K, n_base,      k0, lane);
    v16h b1 = load_frag_b(Bw, N, K, n_base + 16, k0, lane);
    acc[0][0] = wmma_f16(a0, b0, acc[0][0]);
    acc[0][1] = wmma_f16(a0, b1, acc[0][1]);
    acc[1][0] = wmma_f16(a1, b0, acc[1][0]);
    acc[1][1] = wmma_f16(a1, b1, acc[1][1]);
  }

  int mofs = (lane >> 4) << 3;
  int ncol = lane & 15;
#pragma unroll
  for (int mi = 0; mi < 2; ++mi) {
#pragma unroll
    for (int ni = 0; ni < 2; ++ni) {
      int n = n_base + ni * 16 + ncol;
      if (n >= N) continue;
#pragma unroll
      for (int v = 0; v < 8; ++v) {
        int m = m_base + mi * 16 + v + mofs;
        if (m >= M) continue;
        float val = acc[mi][ni][v];
        if (bias) val += bias[n];
        if (residual) val += residual[(size_t)m * N + n];
        if (act == 1) val = gelu_exact(val);
        C[(size_t)m * N + n] = val;
      }
    }
  }
}

// ---------------------------------------------------------------------------
// Elementwise / helper kernels
// ---------------------------------------------------------------------------
// vectorized f32 -> f16: 8 elements per thread (n need not be multiple of 8)
__global__ void k_cvt_f16(const float* __restrict__ s, h16* __restrict__ d, long n) {
  long i8 = ((long)blockIdx.x * blockDim.x + threadIdx.x) * 8;
  if (i8 + 8 <= n) {
    v4f f0 = *(const v4f*)(s + i8);
    v4f f1 = *(const v4f*)(s + i8 + 4);
    v8h o;
#pragma unroll
    for (int k = 0; k < 4; ++k) { o[k] = (h16)f0[k]; o[4 + k] = (h16)f1[k]; }
    *(v8h*)(d + i8) = o;
  } else {
    for (long i = i8; i < n; ++i) d[i] = (h16)s[i];
  }
}

// im2col for circular conv1d (kernel=3, pad=wrap 1): out[b*L+l, j*C+c] = x[b,(l-1+j)%L,c]
__global__ void k_im2col_f16(const float* __restrict__ x, h16* __restrict__ out,
                             int Bn, int L, int C) {
  long total = (long)Bn * L * 3 * C;
  long gid = (long)blockIdx.x * blockDim.x + threadIdx.x;
  if (gid >= total) return;
  int tc  = 3 * C;
  long row = gid / tc;
  int col  = (int)(gid % tc);
  int j = col / C, c = col % C;
  int b = (int)(row / L), l = (int)(row % L);
  int l2 = l - 1 + j; if (l2 < 0) l2 += L; if (l2 >= L) l2 -= L;
  out[gid] = (h16)x[((size_t)b * L + l2) * C + c];
}

// repack conv weight w[O,C,3] -> [O, 3*C] f16 with k = j*C + c
__global__ void k_repack_w_f16(const float* __restrict__ w, h16* __restrict__ out,
                               int O, int C) {
  long total = (long)O * 3 * C;
  long gid = (long)blockIdx.x * blockDim.x + threadIdx.x;
  if (gid >= total) return;
  int tc = 3 * C;
  int o = (int)(gid / tc);
  int col = (int)(gid % tc);
  int j = col / C, c = col % C;
  out[gid] = (h16)w[((size_t)o * C + c) * 3 + j];
}

// y += positional embedding + x_mark @ tw^T + tb   (D = 512, MARK = 4)
__global__ void k_emb_finish(float* __restrict__ y, const float* __restrict__ xm,
                             const float* __restrict__ tw, const float* __restrict__ tb,
                             int Bn, int L) {
  const int D = 512;
  long total = (long)Bn * L * D;
  long gid = (long)blockIdx.x * blockDim.x + threadIdx.x;
  if (gid >= total) return;
  int d = (int)(gid % D);
  long row = gid / D;
  int b = (int)(row / L), l = (int)(row % L);
  int i2 = d >> 1;
  float freq = __expf(-(float)(2 * i2) * (9.21034037197618f / (float)D)); // ln(10000)
  float ang = (float)l * freq;
  float pe = (d & 1) ? cosf(ang) : sinf(ang);
  const float* m = xm + ((size_t)b * L + l) * 4;
  float dot = tb[d];
#pragma unroll
  for (int k = 0; k < 4; ++k) dot += m[k] * tw[(size_t)d * 4 + k];
  y[gid] += pe + dot;
}

// residual-optional LayerNorm over last dim D; one block per row
__global__ void k_layernorm(const float* __restrict__ x, const float* __restrict__ res,
                            const float* __restrict__ g, const float* __restrict__ b,
                            float* __restrict__ y, int D) {
  long row = blockIdx.x;
  const float* xr = x + (size_t)row * D;
  const float* rr = res ? res + (size_t)row * D : nullptr;
  float s = 0.f, s2 = 0.f;
  for (int i = threadIdx.x; i < D; i += blockDim.x) {
    float v = xr[i] + (rr ? rr[i] : 0.f);
    s += v; s2 += v * v;
  }
  __shared__ float sh_s[32], sh_s2[32];
  for (int off = 16; off; off >>= 1) {
    s  += __shfl_down(s, off, 32);
    s2 += __shfl_down(s2, off, 32);
  }
  int wid = threadIdx.x >> 5, lid = threadIdx.x & 31;
  if (lid == 0) { sh_s[wid] = s; sh_s2[wid] = s2; }
  __syncthreads();
  if (threadIdx.x == 0) {
    float ts = 0.f, ts2 = 0.f;
    int nw = blockDim.x >> 5;
    for (int w = 0; w < nw; ++w) { ts += sh_s[w]; ts2 += sh_s2[w]; }
    sh_s[0] = ts; sh_s2[0] = ts2;
  }
  __syncthreads();
  float mean = sh_s[0] / D;
  float var  = sh_s2[0] / D - mean * mean;
  float inv  = rsqrtf(var + 1e-5f);
  for (int i = threadIdx.x; i < D; i += blockDim.x) {
    float v = xr[i] + (rr ? rr[i] : 0.f);
    y[(size_t)row * D + i] = (v - mean) * inv * g[i] + b[i];
  }
}

// ------------------- ProbSparse attention pieces ---------------------------
static __device__ inline unsigned hashu(unsigned x) {
  x ^= x >> 16; x *= 0x7feb352dU; x ^= x >> 15; x *= 0x846ca68bU; x ^= x >> 16;
  return x;
}

// M[l] = max_u q_l . k_{idx(l,u)} - sum_u(.)/L_K   (heads packed in D=512 rows)
__global__ void k_mscore(const float* __restrict__ Q, const float* __restrict__ Km,
                         float* __restrict__ Msc, int Bn, int H, int LQ, int LK,
                         int Upart, unsigned seed) {
  const int D = 512, DH = 64;
  long total = (long)Bn * H * LQ;
  long gid = (long)blockIdx.x * blockDim.x + threadIdx.x;
  if (gid >= total) return;
  int l = (int)(gid % LQ);
  int h = (int)((gid / LQ) % H);
  int b = (int)(gid / ((long)LQ * H));
  const float* q = Q + ((size_t)b * LQ + l) * D + h * DH;
  float mx = -INFINITY, sm = 0.f;
  for (int u = 0; u < Upart; ++u) {
    int kk = (int)(hashu(seed ^ (unsigned)(l * Upart + u)) % (unsigned)LK);
    const float* kr = Km + ((size_t)b * LK + kk) * D + h * DH;
    float acc = 0.f;
    for (int d = 0; d < DH; ++d) acc += q[d] * kr[d];
    mx = fmaxf(mx, acc);
    sm += acc;
  }
  Msc[((size_t)b * H + h) * LQ + l] = mx - sm / (float)LK;
}

// per-(b,h) iterative arg-max top-u (u <= 48, LQ <= 2048); one block per (b,h)
__global__ void k_topk(const float* __restrict__ Msc, int* __restrict__ topi,
                       int LQ, int u) {
  __shared__ float smem[2048];
  __shared__ float rv[256];
  __shared__ int   ri[256];
  int bh = blockIdx.x, tid = threadIdx.x;
  const float* row = Msc + (size_t)bh * LQ;
  for (int i = tid; i < LQ; i += blockDim.x) smem[i] = row[i];
  __syncthreads();
  for (int t = 0; t < u; ++t) {
    float bv = -INFINITY; int bi = 0;
    for (int i = tid; i < LQ; i += blockDim.x)
      if (smem[i] > bv) { bv = smem[i]; bi = i; }
    rv[tid] = bv; ri[tid] = bi;
    __syncthreads();
    for (int s = 128; s > 0; s >>= 1) {
      if (tid < s && rv[tid + s] > rv[tid]) { rv[tid] = rv[tid + s]; ri[tid] = ri[tid + s]; }
      __syncthreads();
    }
    if (tid == 0) { topi[(size_t)bh * u + t] = ri[0]; smem[ri[0]] = -INFINITY; }
    __syncthreads();
  }
}

// vmean[b,h,d] = mean_k V[b,k, h*64+d]; one block (64 threads) per (b,h)
__global__ void k_vmean(const float* __restrict__ V, float* __restrict__ vm,
                        int H, int LK) {
  const int D = 512, DH = 64;
  int bh = blockIdx.x;
  int b = bh / H, h = bh % H;
  int d = threadIdx.x;
  if (d >= DH) return;
  float s = 0.f;
  for (int k = 0; k < LK; ++k) s += V[((size_t)b * LK + k) * D + h * DH + d];
  vm[(size_t)bh * DH + d] = s / (float)LK;
}

// ctx[b,l,:] = vmean broadcast
__global__ void k_ctx_fill(const float* __restrict__ vm, float* __restrict__ ctx,
                           int Bn, int H, int LQ) {
  const int D = 512;
  long total = (long)Bn * LQ * D;
  long gid = (long)blockIdx.x * blockDim.x + threadIdx.x;
  if (gid >= total) return;
  int d = (int)(gid % D);
  long row = gid / D;
  int b = (int)(row / LQ);
  int h = d >> 6;
  ctx[gid] = vm[((size_t)b * H + h) * 64 + (d & 63)];
}

// causal context base: per (b, d-column) running cumsum of V along l (LQ==LK)
__global__ void k_ctx_cumsum(const float* __restrict__ V, float* __restrict__ ctx,
                             int Bn, int LQ) {
  const int D = 512;
  long total = (long)Bn * D;
  long gid = (long)blockIdx.x * blockDim.x + threadIdx.x;
  if (gid >= total) return;
  int b = (int)(gid / D);
  int d = (int)(gid % D);
  float run = 0.f;
  for (int l = 0; l < LQ; ++l) {
    run += V[((size_t)b * LQ + l) * D + d];
    ctx[((size_t)b * LQ + l) * D + d] = run;
  }
}

// one block per (b,h,top-row): scores -> softmax -> upd = attn @ V -> scatter
__global__ void k_attn_update(const float* __restrict__ Q, const float* __restrict__ Km,
                              const float* __restrict__ V, const int* __restrict__ topi,
                              float* __restrict__ ctx, int H, int LQ, int LK,
                              int u, int causal) {
  const int D = 512, DH = 64;
  __shared__ float sc[2048];
  __shared__ float red[256];
  int bh = blockIdx.x / u;
  int ui = blockIdx.x % u;
  int b = bh / H, h = bh % H;
  int lq = topi[(size_t)bh * u + ui];
  int tid = threadIdx.x;
  const float* q = Q + ((size_t)b * LQ + lq) * D + h * DH;

  float lmax = -INFINITY;
  for (int k = tid; k < LK; k += blockDim.x) {
    const float* kr = Km + ((size_t)b * LK + k) * D + h * DH;
    float acc = 0.f;
    for (int d = 0; d < DH; ++d) acc += q[d] * kr[d];
    float s = acc * 0.125f;  // 1/sqrt(64)
    if (causal && k > lq) s = -INFINITY;
    sc[k] = s;
    lmax = fmaxf(lmax, s);
  }
  red[tid] = lmax;
  __syncthreads();
  for (int s = 128; s > 0; s >>= 1) {
    if (tid < s) red[tid] = fmaxf(red[tid], red[tid + s]);
    __syncthreads();
  }
  float mx = red[0];
  __syncthreads();
  float lsum = 0.f;
  for (int k = tid; k < LK; k += blockDim.x) {
    float e = __expf(sc[k] - mx);
    sc[k] = e;
    lsum += e;
  }
  red[tid] = lsum;
  __syncthreads();
  for (int s = 128; s > 0; s >>= 1) {
    if (tid < s) red[tid] += red[tid + s];
    __syncthreads();
  }
  float ssum = red[0];
  __syncthreads();
  if (tid < DH) {
    float acc = 0.f;
    for (int k = 0; k < LK; ++k)
      acc += sc[k] * V[((size_t)b * LK + k) * D + h * DH + tid];
    ctx[((size_t)b * LQ + lq) * D + h * DH + tid] = acc / ssum;
  }
}

// conv-distil post: BN-ish scale, ELU, -inf-padded maxpool(3, stride 2)
__global__ void k_distil_post(const float* __restrict__ src, float* __restrict__ dst,
                              const float* __restrict__ gg, const float* __restrict__ bb,
                              int Bn, int L) {
  const int D = 512;
  int Lout = L / 2;
  long total = (long)Bn * Lout * D;
  long gid = (long)blockIdx.x * blockDim.x + threadIdx.x;
  if (gid >= total) return;
  int d = (int)(gid % D);
  int t = (int)((gid / D) % Lout);
  int b = (int)(gid / ((long)D * Lout));
  const float s = 0.99999500003750f;  // 1/sqrt(1+1e-5)
  float best = -INFINITY;
#pragma unroll
  for (int w = -1; w <= 1; ++w) {
    int p = 2 * t + w;
    if (p < 0 || p >= L) continue;
    float v = src[((size_t)b * L + p) * D + d] * s * gg[d] + bb[d];
    float a = v > 0.f ? v : expm1f(v);
    best = fmaxf(best, a);
  }
  dst[((size_t)b * Lout + t) * D + d] = best;
}

// final slice: out[b, t, c] = proj[(b*1024 + 512 + t), c]
__global__ void k_slice_out(const float* __restrict__ src, float* __restrict__ out) {
  const int CO = 21, PL = 512, LD = 1024;
  long total = (long)4 * PL * CO;
  long gid = (long)blockIdx.x * blockDim.x + threadIdx.x;
  if (gid >= total) return;
  int c = (int)(gid % CO);
  int t = (int)((gid / CO) % PL);
  int b = (int)(gid / ((long)CO * PL));
  out[gid] = src[((size_t)b * LD + PL + t) * CO + c];
}

// ---------------------------------------------------------------------------
// Host orchestration
// ---------------------------------------------------------------------------
struct AttnW { const float *wq,*bq,*wk,*bk,*wv,*bv,*wo,*bo; };

extern "C" void kernel_launch(void* const* d_in, const int* in_sizes, int n_in,
                              void* d_out, int out_size, void* d_ws, size_t ws_size,
                              hipStream_t stream) {
  (void)in_sizes; (void)n_in; (void)out_size; (void)ws_size;
  const int Bn = 4, D = 512, H = 8, DFF = 2048;
  const int Lenc = 2048, Ldec = 1024;
  const int MaxM = Bn * Lenc;

  // ---- inputs (setup_inputs dict order; params as jax pytree: sorted keys) ----
  const float* x_enc      = (const float*)d_in[0];
  const float* x_mark_enc = (const float*)d_in[1];
  const float* x_dec      = (const float*)d_in[2];
  const float* x_mark_dec = (const float*)d_in[3];
  auto P = [&](int i) { return (const float*)d_in[4 + i]; };

  // conv_layers[i]: b, bn_b, bn_g, w  (indices 0..7)
  struct ConvP { const float *b,*bn_b,*bn_g,*w; } conv[2];
  for (int i = 0; i < 2; ++i)
    conv[i] = { P(i*4+0), P(i*4+1), P(i*4+2), P(i*4+3) };
  // dec_emb: tb=8, tok_w=9, tw=10
  const float *dec_tb = P(8), *dec_tokw = P(9), *dec_tw = P(10);
  // dec_layers[j] base = 11 + j*26:
  //  b1,b2, cattn{bk,bo,bq,bv,wk,wo,wq,wv}, n1b,n1g,n2b,n2g,n3b,n3g,
  //  sattn{bk,bo,bq,bv,wk,wo,wq,wv}, w1, w2
  struct DecL { const float *b1,*b2; AttnW ca; const float *n1b,*n1g,*n2b,*n2g,*n3b,*n3g;
                AttnW sa; const float *w1,*w2; } dec[2];
  for (int j = 0; j < 2; ++j) {
    int base = 11 + j * 26;
    dec[j].b1 = P(base+0); dec[j].b2 = P(base+1);
    // pytree order: bk=+2, bo=+3, bq=+4, bv=+5, wk=+6, wo=+7, wq=+8, wv=+9
    dec[j].ca = AttnW{ /*wq*/P(base+8), /*bq*/P(base+4), /*wk*/P(base+6), /*bk*/P(base+2),
                       /*wv*/P(base+9), /*bv*/P(base+5), /*wo*/P(base+7), /*bo*/P(base+3) };
    dec[j].n1b = P(base+10); dec[j].n1g = P(base+11);
    dec[j].n2b = P(base+12); dec[j].n2g = P(base+13);
    dec[j].n3b = P(base+14); dec[j].n3g = P(base+15);
    dec[j].sa = AttnW{ P(base+22), P(base+18), P(base+20), P(base+16),
                       P(base+23), P(base+19), P(base+21), P(base+17) };
    dec[j].w1 = P(base+24); dec[j].w2 = P(base+25);
  }
  const float *dec_nb = P(63), *dec_ng = P(64);
  const float *enc_tb = P(65), *enc_tokw = P(66), *enc_tw = P(67);
  // enc_layers[i] base = 68 + i*16: attn{bk,bo,bq,bv,wk,wo,wq,wv}, b1,b2, n1b,n1g,n2b,n2g, w1,w2
  struct EncL { AttnW a; const float *b1,*b2,*n1b,*n1g,*n2b,*n2g,*w1,*w2; } enc[3];
  for (int i = 0; i < 3; ++i) {
    int base = 68 + i * 16;
    enc[i].a = AttnW{ P(base+6), P(base+2), P(base+4), P(base+0),
                      P(base+7), P(base+3), P(base+5), P(base+1) };
    enc[i].b1 = P(base+8); enc[i].b2 = P(base+9);
    enc[i].n1b = P(base+10); enc[i].n1g = P(base+11);
    enc[i].n2b = P(base+12); enc[i].n2g = P(base+13);
    enc[i].w1 = P(base+14); enc[i].w2 = P(base+15);
  }
  const float *enc_nb = P(116), *enc_ng = P(117);
  const float *proj_b = P(118), *proj_w = P(119);

  // ---- workspace arena ----
  size_t off = 0;
  auto arena = [&](size_t bytes) {
    void* p = (char*)d_ws + off;
    off += (bytes + 255) & ~(size_t)255;
    return p;
  };
  float* xbuf0 = (float*)arena((size_t)MaxM * D * 4);
  float* xbuf1 = (float*)arena((size_t)MaxM * D * 4);
  float* tmp   = (float*)arena((size_t)MaxM * D * 4);
  float* ffh   = (float*)arena((size_t)MaxM * DFF * 4);
  float* qb    = (float*)arena((size_t)MaxM * D * 4);
  float* kb    = (float*)arena((size_t)MaxM * D * 4);
  float* vb    = (float*)arena((size_t)MaxM * D * 4);
  float* ctx   = (float*)arena((size_t)MaxM * D * 4);
  float* encout= (float*)arena((size_t)Bn * 512 * D * 4);
  h16*   A16   = (h16*)arena((size_t)MaxM * DFF * 2);
  h16*   B16   = (h16*)arena((size_t)DFF * D * 2);
  float* vmean = (float*)arena((size_t)Bn * H * 64 * 4);
  float* Msc   = (float*)arena((size_t)Bn * H * Lenc * 4);
  int*   topi  = (int*)arena((size_t)Bn * H * 64 * 4);

  // ---- launch helpers ----
  auto GEMM = [&](const h16* A, const h16* Bw, const float* bias, const float* res,
                  float* C, int M, int N, int K, int act) {
    if ((M % 64 == 0) && (N % 128 == 0) && (K % 32 == 0)) {
      dim3 g((unsigned)(N / 128), (unsigned)(M / 64));
      k_gemm_wmma_fast<<<g, 128, 0, stream>>>(A, Bw, bias, res, C, M, N, K, act);
    } else {
      dim3 g((unsigned)((N + 63) / 64), (unsigned)((M + 63) / 64));
      k_gemm_wmma_gen<<<g, 128, 0, stream>>>(A, Bw, bias, res, C, M, N, K, act);
    }
  };
  auto CVT = [&](const float* s, h16* d, long n) {
    long nb = (n + 7) / 8;
    k_cvt_f16<<<(unsigned)((nb + 255) / 256), 256, 0, stream>>>(s, d, n);
  };
  auto LNORM = [&](const float* x, const float* res, const float* g, const float* b,
                   float* y, int rows) {
    k_layernorm<<<(unsigned)rows, 256, 0, stream>>>(x, res, g, b, y, D);
  };
  auto FFN = [&](const float* xin, int M, const float* w1, const float* b1,
                 const float* w2, const float* b2, float* out) {
    CVT(xin, A16, (long)M * D);
    CVT(w1, B16, (long)DFF * D);
    GEMM(A16, B16, b1, nullptr, ffh, M, DFF, D, /*gelu*/1);
    CVT(ffh, A16, (long)M * DFF);
    CVT(w2, B16, (long)D * DFF);
    GEMM(A16, B16, b2, xin, out, M, D, DFF, 0);
  };
  auto ATTN = [&](const float* xq, int LQ, const float* xkv, int LK,
                  const AttnW& w, int causal, unsigned seed, float* dst) {
    int Mq = Bn * LQ, Mk = Bn * LK;
    CVT(xq, A16, (long)Mq * D);
    CVT(w.wq, B16, (long)D * D);
    GEMM(A16, B16, w.bq, nullptr, qb, Mq, D, D, 0);
    CVT(xkv, A16, (long)Mk * D);
    CVT(w.wk, B16, (long)D * D);
    GEMM(A16, B16, w.bk, nullptr, kb, Mk, D, D, 0);
    CVT(w.wv, B16, (long)D * D);
    GEMM(A16, B16, w.bv, nullptr, vb, Mk, D, D, 0);

    int Up = 5 * (int)::ceil(::log((double)LK)); if (Up > LK) Up = LK;
    int u  = 5 * (int)::ceil(::log((double)LQ)); if (u > LQ)  u = LQ;

    long nms = (long)Bn * H * LQ;
    k_mscore<<<(unsigned)((nms + 255) / 256), 256, 0, stream>>>(qb, kb, Msc, Bn, H, LQ, LK, Up, seed);
    k_topk<<<(unsigned)(Bn * H), 256, 0, stream>>>(Msc, topi, LQ, u);
    if (causal) {
      long nc = (long)Bn * D;
      k_ctx_cumsum<<<(unsigned)((nc + 255) / 256), 256, 0, stream>>>(vb, ctx, Bn, LQ);
    } else {
      k_vmean<<<(unsigned)(Bn * H), 64, 0, stream>>>(vb, vmean, H, LK);
      long nf = (long)Mq * D;
      k_ctx_fill<<<(unsigned)((nf + 255) / 256), 256, 0, stream>>>(vmean, ctx, Bn, H, LQ);
    }
    k_attn_update<<<(unsigned)(Bn * H * u), 256, 0, stream>>>(qb, kb, vb, topi, ctx, H, LQ, LK, u, causal);

    CVT(ctx, A16, (long)Mq * D);
    CVT(w.wo, B16, (long)D * D);
    GEMM(A16, B16, w.bo, /*residual*/xq, dst, Mq, D, D, 0);
  };
  auto EMB = [&](const float* x, const float* xmark, const float* tokw,
                 const float* tw, const float* tb, int L) {
    int M = Bn * L, C = 21;
    long ni = (long)M * 3 * C;
    k_im2col_f16<<<(unsigned)((ni + 255) / 256), 256, 0, stream>>>(x, A16, Bn, L, C);
    long nw = (long)D * 3 * C;
    k_repack_w_f16<<<(unsigned)((nw + 255) / 256), 256, 0, stream>>>(tokw, B16, D, C);
    GEMM(A16, B16, nullptr, nullptr, xbuf0, M, D, 3 * C, 0);
    long nf = (long)M * D;
    k_emb_finish<<<(unsigned)((nf + 255) / 256), 256, 0, stream>>>(xbuf0, xmark, tw, tb, Bn, L);
  };

  // =========================== ENCODER ===========================
  EMB(x_enc, x_mark_enc, enc_tokw, enc_tw, enc_tb, Lenc);
  int L = Lenc;
  for (int i = 0; i < 3; ++i) {
    int M = Bn * L;
    ATTN(xbuf0, L, xbuf0, L, enc[i].a, /*causal*/0, 0x9E3779B9u * (unsigned)(i + 1), tmp);
    LNORM(tmp, nullptr, enc[i].n1g, enc[i].n1b, xbuf1, M);
    FFN(xbuf1, M, enc[i].w1, enc[i].b1, enc[i].w2, enc[i].b2, tmp);
    LNORM(tmp, nullptr, enc[i].n2g, enc[i].n2b, xbuf0, M);
    if (i < 2) {  // conv distil: circular conv (K=1536 GEMM) + BN/ELU/maxpool
      long ni = (long)M * 3 * D;
      k_im2col_f16<<<(unsigned)((ni + 255) / 256), 256, 0, stream>>>(xbuf0, A16, Bn, L, D);
      long nw = (long)D * 3 * D;
      k_repack_w_f16<<<(unsigned)((nw + 255) / 256), 256, 0, stream>>>(conv[i].w, B16, D, D);
      GEMM(A16, B16, conv[i].b, nullptr, tmp, M, D, 3 * D, 0);
      long np = (long)Bn * (L / 2) * D;
      k_distil_post<<<(unsigned)((np + 255) / 256), 256, 0, stream>>>(tmp, xbuf0, conv[i].bn_g, conv[i].bn_b, Bn, L);
      L /= 2;
    }
  }
  LNORM(xbuf0, nullptr, enc_ng, enc_nb, encout, Bn * L);  // L == 512
  const int Lcross = L;

  // =========================== DECODER ===========================
  EMB(x_dec, x_mark_dec, dec_tokw, dec_tw, dec_tb, Ldec);
  int Md = Bn * Ldec;
  for (int j = 0; j < 2; ++j) {
    unsigned s1 = 0x9E3779B9u * (unsigned)(4 + 2 * j);
    unsigned s2 = 0x9E3779B9u * (unsigned)(5 + 2 * j);
    ATTN(xbuf0, Ldec, xbuf0, Ldec, dec[j].sa, /*causal*/1, s1, tmp);
    LNORM(tmp, nullptr, dec[j].n1g, dec[j].n1b, xbuf1, Md);
    ATTN(xbuf1, Ldec, encout, Lcross, dec[j].ca, /*causal*/0, s2, tmp);
    LNORM(tmp, nullptr, dec[j].n2g, dec[j].n2b, xbuf0, Md);
    FFN(xbuf0, Md, dec[j].w1, dec[j].b1, dec[j].w2, dec[j].b2, tmp);
    LNORM(tmp, nullptr, dec[j].n3g, dec[j].n3b, xbuf0, Md);
  }
  LNORM(xbuf0, nullptr, dec_ng, dec_nb, xbuf1, Md);

  // projection to C_OUT=21 and slice of last PRED_LEN rows
  CVT(xbuf1, A16, (long)Md * D);
  CVT(proj_w, B16, (long)21 * D);
  GEMM(A16, B16, proj_b, nullptr, tmp, Md, 21, D, 0);
  long no = (long)Bn * 512 * 21;
  k_slice_out<<<(unsigned)((no + 255) / 256), 256, 0, stream>>>(tmp, (float*)d_out);
}
